// AttentionBlock_19670950215839
// MI455X (gfx1250) — matrix operand order repaired and checked by reading, not compiled, not run on hardware
//
#include <hip/hip_runtime.h>
#include <hip/hip_bf16.h>
#include <math.h>

// ---------------------------------------------------------------------------
// AttentionBlock (GroupNorm -> qkv 1x1 -> flash MHA -> proj 1x1 -> residual)
// B=8, C=512, L=1024, 8 heads, head dim 64, GroupNorm(32).
// bf16 WMMA (v_wmma_f32_16x16x32_bf16), fp32 accumulation.
// Fragment-native layouts: every WMMA operand is loaded with b128 ops.
// GEMMs register-blocked 4 M-tiles x 2 N-tiles per wave (A and B both reused;
// 12 b128 loads feed 8 WMMAs per K-step).
// Attention K/V staged with global_load_async_to_lds_b128 (double-buffered).
// ---------------------------------------------------------------------------

typedef __bf16 bf16;
typedef __attribute__((ext_vector_type(16))) __bf16 v16bf;
typedef __attribute__((ext_vector_type(8)))  __bf16 v8bf;
typedef __attribute__((ext_vector_type(8)))  float  v8f;

#define Bn 8
#define Cn 512
#define Ln 1024
#define HEADS 8
#define CH 64
#define GC 16
#define EPSn 1e-5f
#define QK_SCALE 0.35355339059327379f   // 1/sqrt(sqrt(64))

#define WQF_ELEMS (96 * 16 * 32 * 16)   // qkv weight fragments
#define WPF_ELEMS (32 * 16 * 32 * 16)   // proj weight fragments

__device__ __forceinline__ bf16 f2bf(float f) {
  unsigned u = __builtin_bit_cast(unsigned, f);
  unsigned r = (u + 0x7FFFu + ((u >> 16) & 1u)) >> 16;   // RNE
  return __builtin_bit_cast(bf16, (unsigned short)r);
}

// 16-bit A-fragment (16x32) K index for element j (0..15), lane half `half`
__device__ __forceinline__ int kmapA(int j, int half) {
  return j + (half ? 16 : 0);
}

#define SHUF16(lo, hi) __builtin_shufflevector(lo, hi, 0,1,2,3,4,5,6,7,8,9,10,11,12,13,14,15)

// ---------------------------------------------------------------------------
// Kernel 0: pre-swizzle fp32 weights into bf16 A-fragment lane layout.
// frag[((mt*16+kt)*32 + lane)*16 + j] = W[(mt*16 + lane&15)*512 + kt*32 + kmapA(j, lane>>4)]
// ---------------------------------------------------------------------------
__global__ void prep_w_kernel(const float* __restrict__ qw,
                              const float* __restrict__ pw,
                              bf16* __restrict__ wqf,
                              bf16* __restrict__ wpf) {
  int id = blockIdx.x * 256 + threadIdx.x;
  const float* src;
  bf16* dst;
  int lid;
  if (id < WQF_ELEMS) { src = qw; dst = wqf; lid = id; }
  else                { src = pw; dst = wpf; lid = id - WQF_ELEMS; }
  int j    = lid & 15;
  int lane = (lid >> 4) & 31;
  int kt   = (lid >> 9) & 15;
  int mt   = lid >> 13;
  int half = lane >> 4, lr = lane & 15;
  dst[lid] = f2bf(src[(size_t)(mt * 16 + lr) * Cn + kt * 32 + kmapA(j, half)]);
}

// ---------------------------------------------------------------------------
// Kernel 1: GroupNorm(32) -> bf16 xn_t stored (b, l, c)  [K-contiguous]
// ---------------------------------------------------------------------------
__global__ void gn_kernel(const float* __restrict__ x,
                          const float* __restrict__ gsc,
                          const float* __restrict__ gbi,
                          bf16* __restrict__ xnt) {
  int bg = blockIdx.x;
  int b = bg >> 5, g = bg & 31;
  size_t base = ((size_t)(b * Cn + g * GC)) << 10;
  const float* xp = x + base;

  float s = 0.f, s2 = 0.f;
  for (int i = threadIdx.x; i < GC * Ln; i += 256) {
    float v = xp[i];
    s += v; s2 += v * v;
  }
  __shared__ float rs[256], rs2[256];
  rs[threadIdx.x] = s; rs2[threadIdx.x] = s2;
  __syncthreads();
  for (int o = 128; o > 0; o >>= 1) {
    if (threadIdx.x < o) {
      rs[threadIdx.x]  += rs[threadIdx.x + o];
      rs2[threadIdx.x] += rs2[threadIdx.x + o];
    }
    __syncthreads();
  }
  float mu   = rs[0] * (1.f / (GC * Ln));
  float var  = rs2[0] * (1.f / (GC * Ln)) - mu * mu;
  float rstd = rsqrtf(var + EPSn);

  float sc[GC], bi[GC];
#pragma unroll
  for (int c = 0; c < GC; ++c) {
    sc[c] = gsc[g * GC + c] * rstd;
    bi[c] = gbi[g * GC + c];
  }

  for (int l = threadIdx.x; l < Ln; l += 256) {
    v16bf row;
#pragma unroll
    for (int c = 0; c < GC; ++c)
      row[c] = f2bf((xp[(size_t)c * Ln + l] - mu) * sc[c] + bi[c]);
    *(v16bf*)&xnt[((size_t)(b * Ln) + l) * Cn + g * GC] = row;   // 32B store
  }
}

// ---------------------------------------------------------------------------
// Kernel 2: qkv GEMM, 4 M-tiles (64 out ch = one head/section) x 2 N-tiles.
// q -> qt (head, t, ch), k -> kt (head, s, ch) [scaled], v -> vb (head, ch, s)
// ---------------------------------------------------------------------------
__global__ void __launch_bounds__(256) qkv_kernel(const bf16* __restrict__ wqf,
                                                  const float* __restrict__ bias,
                                                  const bf16* __restrict__ xnt,
                                                  bf16* __restrict__ qt,
                                                  bf16* __restrict__ ktg,
                                                  bf16* __restrict__ vb) {
  int wid  = (blockIdx.x * blockDim.x + threadIdx.x) >> 5;
  int lane = threadIdx.x & 31;
  int half = lane >> 4, lr = lane & 15;

  int nb  = wid & 31;          // 32-column N block
  int rem = wid >> 5;
  int mb  = rem % 24;          // 64-channel output block (one head/section)
  int b   = rem / 24;
  int l0  = nb * 32 + lr;

  const bf16* abase = wqf + (size_t)(mb * 64) * 512 + lane * 16;
  const bf16* brow0 = xnt + ((size_t)(b * Ln) + l0) * Cn + 16 * half;
  const bf16* brow1 = brow0 + 16 * Cn;

  v8f zv = {};
  v8f acc[4][2] = {{zv, zv}, {zv, zv}, {zv, zv}, {zv, zv}};
  for (int kt = 0; kt < 16; ++kt) {
    v16bf Bv0 = *(const v16bf*)(brow0 + kt * 32);
    v16bf Bv1 = *(const v16bf*)(brow1 + kt * 32);
#pragma unroll
    for (int mt = 0; mt < 4; ++mt) {
      v16bf A = *(const v16bf*)(abase + (size_t)(mt * 16 + kt) * 512);
      acc[mt][0] = __builtin_amdgcn_wmma_f32_16x16x32_bf16(false, A, false, Bv0,
                                                           (short)0, acc[mt][0], false, false);
      acc[mt][1] = __builtin_amdgcn_wmma_f32_16x16x32_bf16(false, A, false, Bv1,
                                                           (short)0, acc[mt][1], false, false);
    }
  }

  int sec  = (mb * 64) >> 9;               // 0=q 1=k 2=v (uniform over block)
  int hh   = ((mb * 64) >> 6) & 7;
  int head = b * HEADS + hh;

#pragma unroll
  for (int mt = 0; mt < 4; ++mt) {
    int oc0 = mb * 64 + mt * 16 + 8 * half;   // 8 contiguous out channels
    int cl  = oc0 & 63;
#pragma unroll
    for (int n = 0; n < 2; ++n) {
      int l = l0 + n * 16;
      if (sec < 2) {
        v8bf pack;
#pragma unroll
        for (int r = 0; r < 8; ++r)
          pack[r] = f2bf((acc[mt][n][r] + bias[oc0 + r]) * QK_SCALE);
        bf16* dst = (sec == 0 ? qt : ktg) + ((size_t)head * Ln + l) * CH + cl;
        *(v8bf*)dst = pack;                   // 16B store
      } else {
#pragma unroll
        for (int r = 0; r < 8; ++r)
          vb[((size_t)head * CH + cl + r) * Ln + l] =
              f2bf(acc[mt][n][r] + bias[oc0 + r]);
      }
    }
  }
}

// ---------------------------------------------------------------------------
// Kernel 3: flash attention. 4 waves/block, 16 q-rows each.
// K/V chunks (32 s-positions) staged via global_load_async_to_lds_b128 into
// double-buffered LDS; s_wait_asynccnt + 1 barrier per chunk.
// S^T = K^T Q (softmax row = VGPRs + one xor-16 shuffle); O^T = V * P^T.
// ---------------------------------------------------------------------------
__global__ void __launch_bounds__(128) attn_kernel(const bf16* __restrict__ qt,
                                                   const bf16* __restrict__ ktg,
                                                   const bf16* __restrict__ vb,
                                                   bf16* __restrict__ at) {
  __shared__ __align__(16) bf16 KsL[2][32 * CH];   // (s, c) rows of 128B
  __shared__ __align__(16) bf16 Vs [2][CH * 32];   // (c, s) rows of 64B

  int head = blockIdx.x >> 4;
  int qblk = blockIdx.x & 15;
  int wave = threadIdx.x >> 5;
  int lane = threadIdx.x & 31;
  int half = lane >> 4, lr = lane & 15;
  int tbase = qblk * 64 + wave * 16;

  const bf16* qh = qt  + (size_t)head * Ln * CH;
  const bf16* kh = ktg + (size_t)head * Ln * CH;
  const bf16* vh = vb  + (size_t)head * CH * Ln;

  // Q B-fragments: contiguous ch -> single 32B loads.
  v16bf qf[2];
  qf[0] = *(const v16bf*)(qh + (size_t)(tbase + lr) * CH + 16 * half);
  qf[1] = *(const v16bf*)(qh + (size_t)(tbase + lr) * CH + 32 + 16 * half);

  // Async-stage one 32-wide s chunk (4KB K + 4KB V) into LDS buffer `buf`.
  auto issue_chunk = [&](int s0, int buf) {
    unsigned kl = (unsigned)(size_t)&KsL[buf][0];
    unsigned vl = (unsigned)(size_t)&Vs[buf][0];
    const bf16* kc = kh + (size_t)s0 * CH;          // contiguous 4KB
#pragma unroll
    for (int i = 0; i < 2; ++i) {                   // K: 256 granules / 128 thr
      int g = threadIdx.x * 2 + i;
      unsigned lds = kl + g * 16;
      unsigned long long ga = (unsigned long long)(size_t)(kc + g * 8);
      asm volatile("global_load_async_to_lds_b128 %0, %1, off"
                   :: "v"(lds), "v"(ga) : "memory");
    }
#pragma unroll
    for (int i = 0; i < 2; ++i) {                   // V: 4 granules per c-row
      int g = threadIdx.x * 2 + i;
      int c = g >> 2, qd = g & 3;
      unsigned lds = vl + g * 16;
      unsigned long long ga =
          (unsigned long long)(size_t)(vh + (size_t)c * Ln + s0 + qd * 8);
      asm volatile("global_load_async_to_lds_b128 %0, %1, off"
                   :: "v"(lds), "v"(ga) : "memory");
    }
  };

  issue_chunk(0, 0);

  float mrow = -INFINITY, lsum = 0.f;
  v8f zv = {};
  v8f O[4] = {zv, zv, zv, zv};
  int cur = 0;

  for (int ci = 0; ci < Ln / 32; ++ci) {
    asm volatile("s_wait_asynccnt 0x0" ::: "memory");
    __syncthreads();                                 // chunk `cur` visible
    if (ci + 1 < Ln / 32) issue_chunk((ci + 1) * 32, cur ^ 1);

    const bf16* Kb = &KsL[cur][0];
    const bf16* Vbuf = &Vs[cur][0];

    // S^T tiles (T0: s 0..15, T1: s 16..31); K fragments = 2x ds_load_b128.
    v8f T0 = {}, T1 = {};
#pragma unroll
    for (int f = 0; f < 2; ++f) {
      const bf16* r0 = Kb + (size_t)lr * CH + f * 32 + 16 * half;
      v8bf a0lo = *(const v8bf*)r0;
      v8bf a0hi = *(const v8bf*)(r0 + 8);
      const bf16* r1 = r0 + 16 * CH;
      v8bf a1lo = *(const v8bf*)r1;
      v8bf a1hi = *(const v8bf*)(r1 + 8);
      v16bf A0 = SHUF16(a0lo, a0hi);
      v16bf A1 = SHUF16(a1lo, a1hi);
      T0 = __builtin_amdgcn_wmma_f32_16x16x32_bf16(false, A0, false, qf[f],
                                                   (short)0, T0, false, false);
      T1 = __builtin_amdgcn_wmma_f32_16x16x32_bf16(false, A1, false, qf[f],
                                                   (short)0, T1, false, false);
    }

    // Online softmax (row = fixed t): 16 regs in-lane + xor-16 partner.
    float cmax = -INFINITY;
#pragma unroll
    for (int r = 0; r < 8; ++r) cmax = fmaxf(cmax, fmaxf(T0[r], T1[r]));
    cmax = fmaxf(cmax, __shfl_xor(cmax, 16, 32));
    float mnew  = fmaxf(mrow, cmax);
    float alpha = __expf(mrow - mnew);

    float csum = 0.f;
    v8f P0, P1;
#pragma unroll
    for (int r = 0; r < 8; ++r) {
      float p0 = __expf(T0[r] - mnew);
      float p1 = __expf(T1[r] - mnew);
      P0[r] = p0; P1[r] = p1;
      csum += p0 + p1;
    }
    csum += __shfl_xor(csum, 16, 32);
    lsum = lsum * alpha + csum;
    mrow = mnew;
#pragma unroll
    for (int ct = 0; ct < 4; ++ct)
#pragma unroll
      for (int r = 0; r < 8; ++r) O[ct][r] *= alpha;

    // P^T B-fragment: one xor-16 lane swap.
    v16bf Pb;
#pragma unroll
    for (int r = 0; r < 8; ++r) {
      float own = (lane & 16) ? P1[r] : P0[r];
      float oth = __shfl_xor((lane & 16) ? P0[r] : P1[r], 16, 32);
      Pb[r]     = f2bf((lane & 16) ? oth : own);
      Pb[r + 8] = f2bf((lane & 16) ? own : oth);
    }

    // O^T += V * P^T ; V fragments = 2x ds_load_b128 per c-tile.
#pragma unroll
    for (int ct = 0; ct < 4; ++ct) {
      const bf16* vrow = Vbuf + (size_t)(ct * 16 + lr) * 32 + 16 * half;
      v8bf lo = *(const v8bf*)vrow;
      v8bf hi = *(const v8bf*)(vrow + 8);
      v16bf Va = SHUF16(lo, hi);
      O[ct] = __builtin_amdgcn_wmma_f32_16x16x32_bf16(false, Va, false, Pb,
                                                      (short)0, O[ct], false, false);
    }
    cur ^= 1;
  }

  // Store a^T (b, l, c): 8 contiguous channels per C-fragment -> 16B stores.
  float inv = 1.f / lsum;
  int b = head >> 3, hh = head & 7;
  size_t rowbase = ((size_t)b * Ln + tbase + lr) * Cn + hh * CH + 8 * half;
#pragma unroll
  for (int ct = 0; ct < 4; ++ct) {
    v8bf o;
#pragma unroll
    for (int r = 0; r < 8; ++r) o[r] = f2bf(O[ct][r] * inv);
    *(v8bf*)&at[rowbase + ct * 16] = o;
  }
}

// ---------------------------------------------------------------------------
// Kernel 4: out = x + proj_w @ a + proj_b ; 4 M-tiles x 2 N-tiles per wave.
// ---------------------------------------------------------------------------
__global__ void __launch_bounds__(256) proj_kernel(const bf16* __restrict__ wpf,
                                                   const float* __restrict__ bias,
                                                   const bf16* __restrict__ at,
                                                   const float* __restrict__ x,
                                                   float* __restrict__ out) {
  int wid  = (blockIdx.x * blockDim.x + threadIdx.x) >> 5;
  int lane = threadIdx.x & 31;
  int half = lane >> 4, lr = lane & 15;

  int nb  = wid & 31;          // 32-column N block
  int rem = wid >> 5;
  int mb  = rem & 7;           // 64-channel output block
  int b   = rem >> 3;
  int l0  = nb * 32 + lr;

  const bf16* abase = wpf + (size_t)(mb * 64) * 512 + lane * 16;
  const bf16* brow0 = at + ((size_t)(b * Ln) + l0) * Cn + 16 * half;
  const bf16* brow1 = brow0 + 16 * Cn;

  v8f zv = {};
  v8f acc[4][2] = {{zv, zv}, {zv, zv}, {zv, zv}, {zv, zv}};
  for (int kt = 0; kt < 16; ++kt) {
    v16bf Bv0 = *(const v16bf*)(brow0 + kt * 32);
    v16bf Bv1 = *(const v16bf*)(brow1 + kt * 32);
#pragma unroll
    for (int mt = 0; mt < 4; ++mt) {
      v16bf A = *(const v16bf*)(abase + (size_t)(mt * 16 + kt) * 512);
      acc[mt][0] = __builtin_amdgcn_wmma_f32_16x16x32_bf16(false, A, false, Bv0,
                                                           (short)0, acc[mt][0], false, false);
      acc[mt][1] = __builtin_amdgcn_wmma_f32_16x16x32_bf16(false, A, false, Bv1,
                                                           (short)0, acc[mt][1], false, false);
    }
  }

#pragma unroll
  for (int mt = 0; mt < 4; ++mt) {
    int oc0 = mb * 64 + mt * 16 + 8 * half;
#pragma unroll
    for (int n = 0; n < 2; ++n) {
      int l = l0 + n * 16;
#pragma unroll
      for (int r = 0; r < 8; ++r) {
        size_t idx = (((size_t)(b * Cn + oc0 + r)) << 10) + l;
        out[idx] = acc[mt][n][r] + bias[oc0 + r] + x[idx];
      }
    }
  }
}

// ---------------------------------------------------------------------------
extern "C" void kernel_launch(void* const* d_in, const int* in_sizes, int n_in,
                              void* d_out, int out_size, void* d_ws, size_t ws_size,
                              hipStream_t stream) {
  const float* x    = (const float*)d_in[0];
  const float* gsc  = (const float*)d_in[1];
  const float* gbi  = (const float*)d_in[2];
  const float* qw   = (const float*)d_in[3];
  const float* qbia = (const float*)d_in[4];
  const float* pw   = (const float*)d_in[5];
  const float* pbia = (const float*)d_in[6];
  float* out = (float*)d_out;

  // ws: xn_t 8MB | qt 8MB | kt 8MB | vb 8MB | wqf 1.5MB | wpf 0.5MB  (34MB)
  char* ws = (char*)d_ws;
  bf16* xnt = (bf16*)(ws);
  bf16* qt  = (bf16*)(ws + ((size_t)8  << 20));
  bf16* ktg = (bf16*)(ws + ((size_t)16 << 20));
  bf16* vb  = (bf16*)(ws + ((size_t)24 << 20));
  bf16* wqf = (bf16*)(ws + ((size_t)32 << 20));
  bf16* wpf = (bf16*)(ws + ((size_t)32 << 20) + (size_t)WQF_ELEMS * 2);
  bf16* at  = xnt;   // xn_t dead after qkv GEMM

  prep_w_kernel<<<(WQF_ELEMS + WPF_ELEMS) / 256, 256, 0, stream>>>(qw, pw, wqf, wpf);
  gn_kernel    <<<Bn * 32, 256, 0, stream>>>(x, gsc, gbi, xnt);
  qkv_kernel   <<<Bn * 24 * 32 / 8, 256, 0, stream>>>(wqf, qbia, xnt, qt, ktg, vb);
  attn_kernel  <<<Bn * HEADS * (Ln / 64), 128, 0, stream>>>(qt, ktg, vb, at);
  proj_kernel  <<<Bn * 8 * 32 / 8, 256, 0, stream>>>(wpf, pbia, at, x, out);
}
